// RadonSelfAttention_19267223290564
// MI455X (gfx1250) — matrix operand order repaired and checked by reading, not compiled, not run on hardware
//
#include <hip/hip_runtime.h>

#define PI_F 3.14159265358979f

typedef __attribute__((ext_vector_type(16))) _Float16 v16h;
typedef __attribute__((ext_vector_type(8)))  _Float16 v8h;
typedef __attribute__((ext_vector_type(8)))  float    v8f;

#define IMGN   256
#define ANG    180
#define KP     192          // K / N padded to multiple of 32 / 16
#define NIMG   12           // B*C = 4*3
#define SINO_SZ (IMGN*ANG)
#define SCORE_SZ (IMGN*IMGN)

// ---------------------------------------------------------------------------
// CDNA5 async memory->LDS copy (CDNA5 ISA):
// GLOBAL_LOAD_ASYNC_TO_LDS_B128: copies 16B global->LDS, tracked by ASYNCcnt.
// LDS destination = byte offset (low 32 bits of generic shared pointer).
// ---------------------------------------------------------------------------
static __device__ __forceinline__
void async_lds_copy_b128(const void* gsrc, void* lds_dst) {
    unsigned lofs = (unsigned)(unsigned long long)lds_dst;
    asm volatile("global_load_async_to_lds_b128 %0, %1, off"
                 :: "v"(lofs), "v"(gsrc) : "memory");
}
static __device__ __forceinline__ void wait_async_lds() {
    asm volatile("s_wait_asynccnt 0x0" ::: "memory");
}

// ---------------------------------------------------------------------------
// Branch-free WMMA fragment loaders (CDNA5 16x16x32 f16, CDNA5 ISA)
// A 16x32 (MxK): lane l -> row l%16; halves 0..7 = K kb+(l/16)*8+0..7,
//                                   halves 8..15 = same +16.
// B 32x16 (KxN): lane l -> col l%16; same K mapping as A.
// All operands pre-padded in memory -> contiguous aligned vector loads.
// ---------------------------------------------------------------------------
static __device__ __forceinline__
v16h load_a_h(const _Float16* __restrict__ A, int row, int kb, int lda) {
    const int lane = threadIdx.x & 31;
    const _Float16* p = A + row * lda + kb + ((lane >> 4) << 3);
    v8h lo = *(const v8h*)p;          // 16B
    v8h hi = *(const v8h*)(p + 16);   // 16B
    v16h a;
#pragma unroll
    for (int t = 0; t < 8; ++t) { a[t] = lo[t]; a[t + 8] = hi[t]; }
    return a;
}

static __device__ __forceinline__
v16h load_bT_h(const _Float16* __restrict__ B, int colBase, int kb, int ldb) {
    const int lane = threadIdx.x & 31;
    const _Float16* p = B + (colBase + (lane & 15)) * ldb + kb + ((lane >> 4) << 3);
    v8h lo = *(const v8h*)p;          // 16B
    v8h hi = *(const v8h*)(p + 16);   // 16B
    v16h b;
#pragma unroll
    for (int t = 0; t < 8; ++t) { b[t] = lo[t]; b[t + 8] = hi[t]; }
    return b;
}

// A fragment from f32 source (scores), unconditional vector loads + cvt
static __device__ __forceinline__
v16h load_a_f32(const float* __restrict__ A, int row, int kb, int lda) {
    const int lane = threadIdx.x & 31;
    const float* p = A + row * lda + kb + ((lane >> 4) << 3);
    const float4* q = (const float4*)p;
    float4 x0 = q[0], x1 = q[1];      // K kbase..kbase+7
    float4 x2 = q[4], x3 = q[5];      // K kbase+16..kbase+23
    v16h a;
    a[0]=(_Float16)x0.x; a[1]=(_Float16)x0.y; a[2]=(_Float16)x0.z; a[3]=(_Float16)x0.w;
    a[4]=(_Float16)x1.x; a[5]=(_Float16)x1.y; a[6]=(_Float16)x1.z; a[7]=(_Float16)x1.w;
    a[8]=(_Float16)x2.x; a[9]=(_Float16)x2.y; a[10]=(_Float16)x2.z; a[11]=(_Float16)x2.w;
    a[12]=(_Float16)x3.x; a[13]=(_Float16)x3.y; a[14]=(_Float16)x3.z; a[15]=(_Float16)x3.w;
    return a;
}

// ---------------------------------------------------------------------------
// K0: pack Wq/Wk/Wv into zero-padded f16 [3][192][192]
// ---------------------------------------------------------------------------
__global__ void prep_w_kernel(const float* __restrict__ Wq,
                              const float* __restrict__ Wk,
                              const float* __restrict__ Wv,
                              _Float16* __restrict__ W_h) {
    const int w = blockIdx.x, j = blockIdx.y, k = threadIdx.x;  // k: 0..191
    const float* W = (w == 0) ? Wq : (w == 1) ? Wk : Wv;
    const float v = (j < ANG && k < ANG) ? W[j * ANG + k] : 0.0f;
    W_h[(w * KP + j) * KP + k] = (_Float16)v;
}

// ---------------------------------------------------------------------------
// K1: radon transform -> f16 padded sinogram [12][256][192].
// block=(img, angle-group of 6), 256 threads (detector n).
// Whole 256KB image async-DMA'd into LDS (CDNA5: 320KB/WGP).
// ---------------------------------------------------------------------------
__global__ void radon_kernel(const float* __restrict__ x,
                             _Float16* __restrict__ sino_h) {
    extern __shared__ float img[];            // 256*256 floats = 256 KB
    const int image  = blockIdx.x;            // 0..11
    const int agroup = blockIdx.y;            // 0..29
    const float* src = x + image * (IMGN * IMGN);

    const float4* src4 = (const float4*)src;
    float4*       lds4 = (float4*)img;
    for (int i = threadIdx.x; i < (IMGN * IMGN) / 4; i += blockDim.x)
        async_lds_copy_b128(&src4[i], &lds4[i]);   // global_load_async_to_lds_b128

    const int n = threadIdx.x;                // detector / output column
    if (agroup == 0) {                        // zero the K-pad columns once
#pragma unroll
        for (int a2 = ANG; a2 < KP; ++a2)
            sino_h[(image * IMGN + n) * KP + a2] = (_Float16)0.0f;
    }
    wait_async_lds();
    __syncthreads();

#pragma unroll 1
    for (int aa = 0; aa < 6; ++aa) {
        const int a = agroup * 6 + aa;
        const float th = (float)a * (PI_F / 180.0f);
        const float c = __cosf(th), s = __sinf(th);
        float sx = c * (float)n - 128.0f * (c + s - 1.0f);
        float sy = -s * (float)n - 128.0f * (c - s - 1.0f);
        float acc = 0.0f;
#pragma unroll 4
        for (int y = 0; y < IMGN; ++y) {
            float fx = floorf(sx), fy = floorf(sy);
            int   ix = (int)fx,    iy = (int)fy;
            float wx = sx - fx,    wy = sy - fy;
            bool x0 = (ix >= 0) & (ix < IMGN);
            bool x1 = (ix + 1 >= 0) & (ix + 1 < IMGN);
            bool y0 = (iy >= 0) & (iy < IMGN);
            bool y1 = (iy + 1 >= 0) & (iy + 1 < IMGN);
            float v00 = (x0 & y0) ? img[iy * IMGN + ix]           : 0.0f;
            float v01 = (x1 & y0) ? img[iy * IMGN + ix + 1]       : 0.0f;
            float v10 = (x0 & y1) ? img[(iy + 1) * IMGN + ix]     : 0.0f;
            float v11 = (x1 & y1) ? img[(iy + 1) * IMGN + ix + 1] : 0.0f;
            float top = v00 + wx * (v01 - v00);
            float bot = v10 + wx * (v11 - v10);
            acc += top + wy * (bot - top);
            sx += s; sy += c;
        }
        sino_h[(image * IMGN + n) * KP + a] = (_Float16)acc;
    }
}

// ---------------------------------------------------------------------------
// K2: Q/K = sino @ W^T + b (row-major f16), V stored TRANSPOSED (Vt[192][256])
// so the P@V GEMM gets contiguous B fragments.  All pads zeroed.
// grid=(12 Ntiles, 16 Mtiles, img*3+which), block=32 (one wave).
// ---------------------------------------------------------------------------
__global__ void qkv_kernel(const _Float16* __restrict__ sino_h,
                           const _Float16* __restrict__ W_h,
                           const float* __restrict__ bq,
                           const float* __restrict__ bk,
                           const float* __restrict__ bv,
                           _Float16* __restrict__ Q_h,
                           _Float16* __restrict__ K_h,
                           _Float16* __restrict__ Vt_h) {
    const int tn = blockIdx.x, tm = blockIdx.y;
    const int img = blockIdx.z / 3, which = blockIdx.z % 3;
    const _Float16* A = sino_h + img * IMGN * KP;
    const _Float16* W = W_h + which * KP * KP;
    const float* bias = (which == 0) ? bq : (which == 1) ? bk : bv;
    const int lane = threadIdx.x & 31;
    const int row  = tm * 16 + (lane & 15);

    v8f c = {0.f, 0.f, 0.f, 0.f, 0.f, 0.f, 0.f, 0.f};
#pragma unroll
    for (int kb = 0; kb < KP; kb += 32) {
        v16h a = load_a_h(A, row, kb, KP);
        v16h b = load_bT_h(W, tn * 16, kb, KP);
        c = __builtin_amdgcn_wmma_f32_16x16x32_f16(false, a, false, b,
                                                   (short)0, c, false, false);
    }
    const int col  = tn * 16 + (lane & 15);
    const int mrow = tm * 16 + ((lane >> 4) << 3);
    const bool live = (col < ANG);
    const float bb = live ? bias[col] : 0.0f;
    if (which == 2) {
#pragma unroll
        for (int r = 0; r < 8; ++r) {
            const float v = live ? (c[r] + bb) : 0.0f;
            Vt_h[(img * KP + col) * IMGN + (mrow + r)] = (_Float16)v;
        }
    } else {
        _Float16* O = ((which == 0) ? Q_h : K_h) + img * IMGN * KP;
#pragma unroll
        for (int r = 0; r < 8; ++r) {
            const float v = live ? (c[r] + bb) : 0.0f;
            O[(mrow + r) * KP + col] = (_Float16)v;
        }
    }
}

// ---------------------------------------------------------------------------
// K3: logits = Q @ K^T / sqrt(180)  (WMMA, M=256 N=256 K=192, pads are zero)
// ---------------------------------------------------------------------------
__global__ void logits_kernel(const _Float16* __restrict__ Q_h,
                              const _Float16* __restrict__ K_h,
                              float* __restrict__ S) {
    const int tn = blockIdx.x, tm = blockIdx.y, img = blockIdx.z;
    const _Float16* Aq = Q_h + img * IMGN * KP;
    const _Float16* Bk = K_h + img * IMGN * KP;
    float*          So = S   + img * SCORE_SZ;
    const int lane = threadIdx.x & 31;
    const int row  = tm * 16 + (lane & 15);

    v8f c = {0.f, 0.f, 0.f, 0.f, 0.f, 0.f, 0.f, 0.f};
#pragma unroll
    for (int kb = 0; kb < KP; kb += 32) {
        v16h a = load_a_h(Aq, row, kb, KP);
        v16h b = load_bT_h(Bk, tn * 16, kb, KP);
        c = __builtin_amdgcn_wmma_f32_16x16x32_f16(false, a, false, b,
                                                   (short)0, c, false, false);
    }
    const float scale = 0.07453559925f;   // 1/sqrt(180)
    const int col  = tn * 16 + (lane & 15);
    const int mrow = tm * 16 + ((lane >> 4) << 3);
#pragma unroll
    for (int r = 0; r < 8; ++r)
        So[(mrow + r) * IMGN + col] = c[r] * scale;
}

// ---------------------------------------------------------------------------
// K4: row softmax over 256 columns.  block = one row, 256 threads.
// ---------------------------------------------------------------------------
__global__ void softmax_kernel(float* __restrict__ S) {
    __shared__ float red[256];
    float* row = S + blockIdx.x * IMGN;
    const int t = threadIdx.x;
    float v = row[t];
    red[t] = v; __syncthreads();
    for (int off = 128; off > 0; off >>= 1) {
        if (t < off) red[t] = fmaxf(red[t], red[t + off]);
        __syncthreads();
    }
    const float m = red[0]; __syncthreads();
    const float e = __expf(v - m);
    red[t] = e; __syncthreads();
    for (int off = 128; off > 0; off >>= 1) {
        if (t < off) red[t] = red[t] + red[t + off];
        __syncthreads();
    }
    row[t] = e / red[0];
}

// ---------------------------------------------------------------------------
// K5: att = P @ V  (WMMA, M=256 N=192 K=256; B = Vt rows, contiguous)
// ---------------------------------------------------------------------------
__global__ void attv_kernel(const float* __restrict__ S,
                            const _Float16* __restrict__ Vt_h,
                            float* __restrict__ att) {
    const int tn = blockIdx.x, tm = blockIdx.y, img = blockIdx.z;
    const float*    As = S    + img * SCORE_SZ;
    const _Float16* Bv = Vt_h + img * KP * IMGN;
    float*          O  = att  + img * SINO_SZ;
    const int lane = threadIdx.x & 31;
    const int row  = tm * 16 + (lane & 15);

    v8f c = {0.f, 0.f, 0.f, 0.f, 0.f, 0.f, 0.f, 0.f};
#pragma unroll
    for (int kb = 0; kb < IMGN; kb += 32) {
        v16h a = load_a_f32(As, row, kb, IMGN);
        v16h b = load_bT_h(Bv, tn * 16, kb, IMGN);
        c = __builtin_amdgcn_wmma_f32_16x16x32_f16(false, a, false, b,
                                                   (short)0, c, false, false);
    }
    const int col = tn * 16 + (lane & 15);
    if (col < ANG) {
        const int mrow = tm * 16 + ((lane >> 4) << 3);
#pragma unroll
        for (int r = 0; r < 8; ++r)
            O[(mrow + r) * ANG + col] = c[r];
    }
}

// ---------------------------------------------------------------------------
// K6: ramp filter as exact circular convolution (ifft(ramp) == 2*f):
//   h(0)=0.5, h(d odd)=-2/(pi*min(d,512-d))^2, h(even)=0.
// ---------------------------------------------------------------------------
__global__ void rampfilter_kernel(const float* __restrict__ att,
                                  float* __restrict__ filt) {
    const int img = blockIdx.x, n = blockIdx.y, a = threadIdx.x;
    if (a >= ANG) return;
    const float* A = att + img * SINO_SZ;
    float acc = 0.0f;
#pragma unroll 1
    for (int m = 0; m < IMGN; ++m) {
        const int d = (n - m) & 511;
        float w;
        if (d == 0) {
            w = 0.5f;
        } else if (d & 1) {
            const int dm = (d < 256) ? d : 512 - d;
            w = -2.0f / (PI_F * PI_F * (float)dm * (float)dm);
        } else {
            continue;
        }
        acc += w * A[m * ANG + a];
    }
    filt[(img * IMGN + n) * ANG + a] = acc;
}

// ---------------------------------------------------------------------------
// K7: backprojection + circle mask + pi/(2A) scale.
// block=(img, rowgroup of 16), 256 threads; filtered sinogram (184KB)
// async-DMA'd into LDS.
// ---------------------------------------------------------------------------
__global__ void backproj_kernel(const float* __restrict__ filt,
                                float* __restrict__ out) {
    extern __shared__ float sf[];             // 256*180 floats
    __shared__ float sc[ANG], ss[ANG];
    const int img = blockIdx.x, rg = blockIdx.y;
    const float4* F4 = (const float4*)(filt + img * SINO_SZ);
    float4* sf4 = (float4*)sf;
    for (int i = threadIdx.x; i < SINO_SZ / 4; i += blockDim.x)
        async_lds_copy_b128(&F4[i], &sf4[i]);  // global_load_async_to_lds_b128
    if (threadIdx.x < ANG) {
        const float th = (float)threadIdx.x * (PI_F / 180.0f);
        sc[threadIdx.x] = __cosf(th);
        ss[threadIdx.x] = __sinf(th);
    }
    wait_async_lds();
    __syncthreads();

    const int j = threadIdx.x;
    const float yr = (float)j - 128.0f;       // reference ypr (2nd index)
#pragma unroll 1
    for (int p = 0; p < 16; ++p) {
        const int i = rg * 16 + p;
        const float xr = (float)i - 128.0f;   // reference xpr (1st index)
        float acc = 0.0f;
#pragma unroll 4
        for (int a = 0; a < ANG; ++a) {
            const float t  = yr * sc[a] - xr * ss[a] + 128.0f;
            const float ft = floorf(t);
            const int   i0 = (int)ft;
            const float fr = t - ft;
            const int c0 = min(max(i0, 0), IMGN - 1);
            const int c1 = min(max(i0 + 1, 0), IMGN - 1);
            const float v = (1.0f - fr) * sf[c0 * ANG + a] + fr * sf[c1 * ANG + a];
            acc += (t >= 0.0f && t <= 255.0f) ? v : 0.0f;
        }
        const float mask = (xr * xr + yr * yr <= 128.0f * 128.0f) ? 1.0f : 0.0f;
        out[(img * IMGN + i) * IMGN + j] = acc * mask * (PI_F / 360.0f);
    }
}

// ---------------------------------------------------------------------------
extern "C" void kernel_launch(void* const* d_in, const int* in_sizes, int n_in,
                              void* d_out, int out_size, void* d_ws, size_t ws_size,
                              hipStream_t stream) {
    const float* x  = (const float*)d_in[0];
    const float* Wq = (const float*)d_in[1];
    const float* bq = (const float*)d_in[2];
    const float* Wk = (const float*)d_in[3];
    const float* bk = (const float*)d_in[4];
    const float* Wv = (const float*)d_in[5];
    const float* bv = (const float*)d_in[6];
    float* out = (float*)d_out;

    char* p = (char*)d_ws;                                   // all chunks 512B-multiples
    _Float16* sino_h = (_Float16*)p;  p += (size_t)NIMG * IMGN * KP * 2;
    _Float16* W_h    = (_Float16*)p;  p += (size_t)3 * KP * KP * 2;
    _Float16* Q_h    = (_Float16*)p;  p += (size_t)NIMG * IMGN * KP * 2;
    _Float16* K_h    = (_Float16*)p;  p += (size_t)NIMG * IMGN * KP * 2;
    _Float16* Vt_h   = (_Float16*)p;  p += (size_t)NIMG * KP * IMGN * 2;
    float*    scores = (float*)p;     p += (size_t)NIMG * SCORE_SZ * 4;
    float*    att    = (float*)p;     p += (size_t)NIMG * SINO_SZ * 4;
    float*    filt   = (float*)p;     // + NIMG*SINO_SZ*4  (~11.9 MiB total)

    // K0: pack weights to padded f16
    prep_w_kernel<<<dim3(3, KP), KP, 0, stream>>>(Wq, Wk, Wv, W_h);
    // K1: radon (256KB dynamic LDS / block, async DMA fill)
    radon_kernel<<<dim3(NIMG, 30), 256, IMGN * IMGN * sizeof(float), stream>>>(x, sino_h);
    // K2: QKV projections (WMMA)
    qkv_kernel<<<dim3(12, 16, NIMG * 3), 32, 0, stream>>>(sino_h, W_h, bq, bk, bv,
                                                          Q_h, K_h, Vt_h);
    // K3: logits (WMMA)
    logits_kernel<<<dim3(16, 16, NIMG), 32, 0, stream>>>(Q_h, K_h, scores);
    // K4: softmax rows
    softmax_kernel<<<NIMG * IMGN, 256, 0, stream>>>(scores);
    // K5: att = P @ V (WMMA)
    attv_kernel<<<dim3(12, 16, NIMG), 32, 0, stream>>>(scores, Vt_h, att);
    // K6: ramp filter (closed-form circular convolution)
    rampfilter_kernel<<<dim3(NIMG, IMGN), 256, 0, stream>>>(att, filt);
    // K7: backprojection (184KB dynamic LDS / block, async DMA fill)
    backproj_kernel<<<dim3(NIMG, 16), 256, SINO_SZ * sizeof(float), stream>>>(filt, out);
}
